// HGC_Layer_80178449482126
// MI455X (gfx1250) — matrix operand order repaired and checked by reading, not compiled, not run on hardware
//
#include <hip/hip_runtime.h>
#include <stdint.h>

// ---------------- CDNA5 types ----------------
typedef __attribute__((ext_vector_type(16))) __bf16 v16bf;
typedef __attribute__((ext_vector_type(2)))  __bf16 v2bf;
typedef __attribute__((ext_vector_type(2)))  float  v2f;
typedef __attribute__((ext_vector_type(8)))  float  v8f;
typedef __attribute__((ext_vector_type(4)))  unsigned int u32x4;
typedef __attribute__((ext_vector_type(8)))  int    i32x8;
typedef __attribute__((ext_vector_type(4)))  int    i32x4;

#if defined(__has_builtin)
#if __has_builtin(__builtin_amdgcn_tensor_load_to_lds) && \
    __has_builtin(__builtin_amdgcn_s_wait_tensorcnt)
#define HAVE_TDM 1
#endif
#if __has_builtin(__builtin_amdgcn_cvt_pk_bf16_f32)
#define HAVE_PK_BF16 1
#endif
#endif

// Problem sizes (B=4, S=4096, H=1024, D=2048)
constexpr int TOKS = 16384;   // B*S  (GEMM M)
constexpr int KH   = 1024;    // H    (GEMM K of main GEMM)
constexpr int ND   = 2048;    // D    (GEMM N)

// ---------------- helpers ----------------
__device__ __forceinline__ uint32_t pk2(float a, float b) {
#if defined(HAVE_PK_BF16)
  union { v2bf v; uint32_t u; } c;
  c.v = __builtin_amdgcn_cvt_pk_bf16_f32(a, b);
  return c.u;
#else
  // fptrunc <2 x float> -> <2 x bfloat>: selects the native packed convert
  // on gfx1250; falls back to inline RNE expansion otherwise.
  v2f x; x[0] = a; x[1] = b;
  union { v2bf v; uint32_t u; } c;
  c.v = __builtin_convertvector(x, v2bf);
  return c.u;
#endif
}
__device__ __forceinline__ void cvt_store16(char* dst, const float* f) {
  uint4 q0, q1;
  q0.x = pk2(f[0], f[1]);  q0.y = pk2(f[2], f[3]);
  q0.z = pk2(f[4], f[5]);  q0.w = pk2(f[6], f[7]);
  q1.x = pk2(f[8], f[9]);  q1.y = pk2(f[10], f[11]);
  q1.z = pk2(f[12], f[13]); q1.w = pk2(f[14], f[15]);
  *(uint4*)dst        = q0;
  *(uint4*)(dst + 16) = q1;
}
__device__ __forceinline__ void load16f(float* f, const float* g) {
  float4 t0 = *(const float4*)(g);
  float4 t1 = *(const float4*)(g + 4);
  float4 t2 = *(const float4*)(g + 8);
  float4 t3 = *(const float4*)(g + 12);
  f[0]=t0.x; f[1]=t0.y; f[2]=t0.z; f[3]=t0.w;
  f[4]=t1.x; f[5]=t1.y; f[6]=t1.z; f[7]=t1.w;
  f[8]=t2.x; f[9]=t2.y; f[10]=t2.z; f[11]=t2.w;
  f[12]=t3.x; f[13]=t3.y; f[14]=t3.z; f[15]=t3.w;
}

union AB { uint4 q[2]; v16bf v; };

// LDS tiles: 128 rows x 32 bf16, row stride 80 B (64 B data + 16 B pad,
// conflict-free b128 fetches: 80 B = 20 banks, gcd(20,64)=4 -> full spread).
// A-operand lane (L + 16*h): K-octets at row*80 + 16h and +32 (ISA 7.12.2).
// B-operand lane (L + 16*h): K-octets at row*80 + 16h and +32 (same as A).
__device__ __forceinline__ void wmma_step(const char* smA, const char* smB,
                                          int L, int h, int wrow, int wcol,
                                          v8f acc[2][4]) {
  AB a[2];
#pragma unroll
  for (int mt = 0; mt < 2; ++mt) {
    const char* p = smA + (wrow * 32 + mt * 16 + L) * 80 + h * 16;
    a[mt].q[0] = *(const uint4*)(p);
    a[mt].q[1] = *(const uint4*)(p + 32);
  }
#pragma unroll
  for (int nt = 0; nt < 4; ++nt) {
    const char* p = smB + (wcol * 64 + nt * 16 + L) * 80 + h * 16;
    AB b;
    b.q[0] = *(const uint4*)(p);
    b.q[1] = *(const uint4*)(p + 32);
#pragma unroll
    for (int mt = 0; mt < 2; ++mt)
      acc[mt][nt] = __builtin_amdgcn_wmma_f32_16x16x32_bf16(
          false, a[mt].v, false, b.v, (short)0, acc[mt][nt], false, false);
  }
}

// ---------------- bias: bp[n] = sum_m b_q[m] * hkm[(m+n) & 2047] ----------------
__global__ __launch_bounds__(256) void hgc_bias_kernel(
    const float* __restrict__ bq, const float* __restrict__ hkm,
    float* __restrict__ bp) {
  const int n = blockIdx.x * 256 + threadIdx.x;
  float acc = 0.f;
  for (int m = 0; m < ND; ++m) acc += bq[m] * hkm[(m + n) & (ND - 1)];
  bp[n] = acc;
}

// ---------------- prep: W'^T[n][h] = sum_m hkm[(n+m)%D] * W_q[h][m] ----------------
// GEMM: M-dim = n (2048), N-dim = h (1024), K = m (2048).
// A = circulant tile generated from hkm (held duplicated in LDS, no wrap needed),
// B = W_q rows (contiguous fp32 in global along m), output bf16 transposed.
__global__ __launch_bounds__(256) void hgc_prep_kernel(
    const float* __restrict__ Wq,      // [1024][2048] fp32
    const float* __restrict__ hkm,     // [2048] fp32
    uint16_t* __restrict__ WpT) {      // [2048][1024] bf16 out
  __shared__ __align__(16) float hkm_s[4128];   // hkm duplicated
  __shared__ __align__(16) char  smA[128 * 80];
  __shared__ __align__(16) char  smB[128 * 80];

  const int tid  = threadIdx.x;
  const int lane = tid & 31, wave = tid >> 5;
  const int L = lane & 15, h = lane >> 4;
  const int wrow = wave & 3, wcol = wave >> 2;
  const int h0 = blockIdx.x * 128;   // N-dim (columns, H)
  const int n0 = blockIdx.y * 128;   // M-dim (rows of W'^T)

  for (int i = tid; i < 4128; i += 256) hkm_s[i] = hkm[i & (ND - 1)];
  __syncthreads();

  const v8f zero = {0.f,0.f,0.f,0.f,0.f,0.f,0.f,0.f};
  v8f acc[2][4];
#pragma unroll
  for (int mt = 0; mt < 2; ++mt)
#pragma unroll
    for (int nt = 0; nt < 4; ++nt) acc[mt][nt] = zero;

  const int rowS = tid >> 1, seg = tid & 1;        // staging: 2 threads per row
  for (int m0 = 0; m0 < ND; m0 += 32) {
    // A tile: C[n0+row, m0+j] = hkm[(n0+row+m0+j) mod D] -> contiguous in hkm_s
    {
      const int base = n0 + rowS + m0 + seg * 16;  // always < 4128
      float f[16];
#pragma unroll
      for (int i = 0; i < 16; ++i) f[i] = hkm_s[base + i];
      cvt_store16(smA + rowS * 80 + seg * 32, f);
    }
    // B tile: W_q[h0+row][m0 + seg*16 .. +16) (contiguous fp32)
    {
      float f[16];
      load16f(f, Wq + (size_t)(h0 + rowS) * ND + m0 + seg * 16);
      cvt_store16(smB + rowS * 80 + seg * 32, f);
    }
    __syncthreads();
    wmma_step(smA, smB, L, h, wrow, wcol, acc);
    __syncthreads();
  }
  // store D (f32 16x16 layout: lane holds N=L, rows 8h+v) as bf16, transposed target
#pragma unroll
  for (int nt = 0; nt < 4; ++nt) {
    const int col = h0 + wcol * 64 + nt * 16 + L;
#pragma unroll
    for (int mt = 0; mt < 2; ++mt) {
      const int rowb = n0 + wrow * 32 + mt * 16 + 8 * h;
#pragma unroll
      for (int v = 0; v < 8; ++v)
        WpT[(size_t)(rowb + v) * KH + col] =
            (uint16_t)(pk2(acc[mt][nt][v], 0.f) & 0xFFFFu);
    }
  }
}

// ---------------- main: out[m][n] = sum_k X[m][k] * W'^T[n][k] + bp[n] ----------------
// Double-buffered pipeline: TDM(i+1) + X-loads(i+1) in flight during WMMA(i).
__global__ __launch_bounds__(256) void hgc_main_kernel(
    const float*    __restrict__ X,    // [16384][1024] fp32
    const uint16_t* __restrict__ WpT,  // [2048][1024] bf16
    const float*    __restrict__ bp,   // [2048]
    float*          __restrict__ out) { // [16384][2048] fp32
  __shared__ __align__(16) char smA[2][128 * 80];
  __shared__ __align__(16) char smB[2][128 * 80];

  const int tid  = threadIdx.x;
  const int lane = tid & 31, wave = tid >> 5;
  const int L = lane & 15, h = lane >> 4;
  const int wrow = wave & 3, wcol = wave >> 2;
  const int n0 = blockIdx.x * 128;
  const int m0 = blockIdx.y * 128;

  const v8f zero = {0.f,0.f,0.f,0.f,0.f,0.f,0.f,0.f};
  v8f acc[2][4];
#pragma unroll
  for (int mt = 0; mt < 2; ++mt)
#pragma unroll
    for (int nt = 0; nt < 4; ++nt) acc[mt][nt] = zero;

  const int rowS = tid >> 1, seg = tid & 1;
  const float* xrow = X + (size_t)(m0 + rowS) * KH + seg * 16;
  constexpr int NSTEP = KH / 32;

#if defined(HAVE_TDM)
  // Tensor DMA descriptor (D#, ISA 08_async_tensor §8): 2D tensor W'^T
  // [2048 x 1024] bf16, tile 128(rows) x 32(k), LDS pad 16 B per 64 B row
  // (pad_interval=3 -> 16 DW, pad_amount=3 -> 4 DW) => 80 B LDS row stride.
  const unsigned ldsB0 = (unsigned)(uintptr_t)(void*)&smB[0][0];
  const unsigned ldsB1 = (unsigned)(uintptr_t)(void*)&smB[1][0];
  const unsigned long long gbase =
      (unsigned long long)(uintptr_t)WpT + (unsigned long long)n0 * KH * 2ull;
  i32x8 g1;
  g1[0] = (int)((1u << 16) | (1u << 20) | (3u << 22) | (3u << 25)); // dsz=2B,pad
  g1[1] = (int)(((unsigned)KH & 0xFFFFu) << 16);                    // dim0 lo16
  g1[2] = (int)(((unsigned)KH >> 16) | (((unsigned)ND & 0xFFFFu) << 16));
  g1[3] = (int)(((unsigned)ND >> 16) | (32u << 16));                // tile_dim0=32
  g1[4] = (int)(128u);                                              // tile_dim1=128
  g1[5] = (int)KH;                                                  // dim0_stride
  g1[6] = 0;
  g1[7] = 0;
  const i32x4 gz4 = {0, 0, 0, 0};                    // D# groups 2/3 (2D: unused)
  const i32x8 gz8 = {0, 0, 0, 0, 0, 0, 0, 0};        // trailing operand (6-arg)
#define TDM_ISSUE(step, ldsaddr)                                         \
  do {                                                                   \
    const unsigned long long ga = gbase + (unsigned)(step) * 64u;        \
    u32x4 g0;                                                            \
    g0[0] = 1u;                       /* count=1 (valid descriptor) */   \
    g0[1] = (ldsaddr);                /* lds_addr */                     \
    g0[2] = (unsigned)ga;             /* global_addr[31:0] */            \
    g0[3] = (unsigned)(ga >> 32) | (2u << 30); /* addr[56:32]|type=2 */  \
    __builtin_amdgcn_tensor_load_to_lds(g0, g1, gz4, gz4, gz8, 0);       \
  } while (0)
#endif

  // ---- prologue: stage step 0 into buffer 0 ----
  float fr[16];
  load16f(fr, xrow);
  cvt_store16(smA[0] + rowS * 80 + seg * 32, fr);
#if defined(HAVE_TDM)
  if (wave == 0) TDM_ISSUE(0, ldsB0);
#else
  {
    const uint16_t* g = WpT + (size_t)(n0 + rowS) * KH + seg * 16;
    char* p = smB[0] + rowS * 80 + seg * 32;
    *(uint4*)p        = *(const uint4*)(g);
    *(uint4*)(p + 16) = *(const uint4*)(g + 8);
  }
#endif

  for (int i = 0; i < NSTEP; ++i) {
    const int cur = i & 1, nxt = cur ^ 1;
#if defined(HAVE_TDM)
    __builtin_amdgcn_s_wait_tensorcnt(0);   // TDM(i) landed (no-op for waves 1..7)
#endif
    __syncthreads();                        // bufs[cur] ready; bufs[nxt] free
    const bool more = (i + 1 < NSTEP);
    if (more) {
      load16f(fr, xrow + (i + 1) * 32);     // X loads overlap with WMMA below
#if defined(HAVE_TDM)
      if (wave == 0) TDM_ISSUE(i + 1, nxt ? ldsB1 : ldsB0);  // DMA overlaps WMMA
#else
      const uint16_t* g = WpT + (size_t)(n0 + rowS) * KH + (i + 1) * 32 + seg * 16;
      char* p = smB[nxt] + rowS * 80 + seg * 32;
      *(uint4*)p        = *(const uint4*)(g);
      *(uint4*)(p + 16) = *(const uint4*)(g + 8);
#endif
    }
    wmma_step(smA[cur], smB[cur], L, h, wrow, wcol, acc);
    if (more) cvt_store16(smA[nxt] + rowS * 80 + seg * 32, fr);
  }

  // epilogue: add bias, fp32 stores (lanes 0..15 of a row -> 64 B segments)
#pragma unroll
  for (int nt = 0; nt < 4; ++nt) {
    const int col = n0 + wcol * 64 + nt * 16 + L;
    const float bias = bp[col];
#pragma unroll
    for (int mt = 0; mt < 2; ++mt) {
      const int rowb = m0 + wrow * 32 + mt * 16 + 8 * h;
#pragma unroll
      for (int v = 0; v < 8; ++v)
        out[(size_t)(rowb + v) * ND + col] = acc[mt][nt][v] + bias;
    }
  }
}

// ---------------- launch ----------------
extern "C" void kernel_launch(void* const* d_in, const int* in_sizes, int n_in,
                              void* d_out, int out_size, void* d_ws, size_t ws_size,
                              hipStream_t stream) {
  const float* X   = (const float*)d_in[0];   // hidden_states [4,4096,1024]
  const float* Wq  = (const float*)d_in[1];   // [1024,2048]
  const float* bq  = (const float*)d_in[2];   // [2048]
  const float* hkm = (const float*)d_in[3];   // [1,2048]
  float* out = (float*)d_out;                 // [4,4096,2048] fp32

  uint16_t* WpT = (uint16_t*)d_ws;                              // 4 MB bf16
  float*    bp  = (float*)((char*)d_ws + (size_t)ND * KH * 2);  // 8 KB

  hgc_bias_kernel<<<ND / 256, 256, 0, stream>>>(bq, hkm, bp);
  hgc_prep_kernel<<<dim3(KH / 128, ND / 128), 256, 0, stream>>>(Wq, hkm, WpT);
  hgc_main_kernel<<<dim3(ND / 128, TOKS / 128), 256, 0, stream>>>(X, WpT, bp, out);
}
